// djmodel_60962765799799
// MI455X (gfx1250) — compile-verified, matrix operand order checked by reading, not run on hardware
//
#include <hip/hip_runtime.h>
#include <hip/hip_bf16.h>
#include <math.h>

// ---------------------------------------------------------------------------
// CDNA5 (gfx1250) wave32 WMMA types
// ---------------------------------------------------------------------------
typedef __attribute__((ext_vector_type(16))) __bf16 bf16x16;
typedef __attribute__((ext_vector_type(8)))  float  v8f;
typedef __attribute__((ext_vector_type(4)))  int    i32x4;

// Problem constants: B=128, T=256, I=512, H=1024
#define ROWS   32768     // B*T
#define DIM_I  512
#define DIM_H  1024
#define DIM_G  4096      // 4H
#define DIM_T  256
#define DIM_B  128
#define LDA    40        // LDS row stride (ushort): 80B -> 16B aligned, conflict-free

// ---------------------------------------------------------------------------
// CDNA5 async global->LDS copy (ASYNCcnt) with synchronous fallback
// ---------------------------------------------------------------------------
#if defined(__HIP_DEVICE_COMPILE__) && defined(__has_builtin)
#if __has_builtin(__builtin_amdgcn_global_load_async_to_lds_b128) && \
    __has_builtin(__builtin_amdgcn_s_wait_asynccnt)
#define USE_ASYNC 1
#endif
#endif

__device__ __forceinline__ void cp_b128(const unsigned short* g, unsigned short* l) {
#ifdef USE_ASYNC
    __builtin_amdgcn_global_load_async_to_lds_b128(
        (__attribute__((address_space(1))) i32x4*)g,
        (__attribute__((address_space(3))) i32x4*)l,
        0, 0);
#else
    *(uint4*)l = *(const uint4*)g;
#endif
}

#ifdef USE_ASYNC
#define TILE_WAIT(n) __builtin_amdgcn_s_wait_asynccnt(n)
#else
#define TILE_WAIT(n) do {} while (0)
#endif

// ---------------------------------------------------------------------------
// Fast activation helpers (native v_tanh_f32 / v_exp_f32 / v_rcp_f32 if present)
// ---------------------------------------------------------------------------
#if defined(__HIP_DEVICE_COMPILE__) && __has_builtin(__builtin_amdgcn_tanhf)
#define TANHF(x) __builtin_amdgcn_tanhf(x)
#else
#define TANHF(x) tanhf(x)
#endif

__device__ __forceinline__ float sigmoidf_(float x) {
#if defined(__HIP_DEVICE_COMPILE__) && __has_builtin(__builtin_amdgcn_exp2f) && \
    __has_builtin(__builtin_amdgcn_rcpf)
    float e = __builtin_amdgcn_exp2f(x * -1.44269504088896340736f);
    return __builtin_amdgcn_rcpf(1.0f + e);
#else
    return 1.0f / (1.0f + __expf(-x));
#endif
}

__device__ __forceinline__ unsigned short f2bf(float x) {
    union { float f; unsigned int u; } v; v.f = x;
    unsigned int u = v.u;
    u += 0x7FFFu + ((u >> 16) & 1u);      // round-to-nearest-even
    return (unsigned short)(u >> 16);
}

// ISA 7.12.2 16-bit A/B fragment: lanes 0-15 hold K {0..7,16..23},
// lanes 16-31 hold K {8..15,24..31}; two ds_load_b128 per fragment.
__device__ __forceinline__ bf16x16 load_frag(const unsigned short* rowbase, int kb) {
    bf16x16 r;
    ((uint4*)&r)[0] = *(const uint4*)(rowbase + kb);
    ((uint4*)&r)[1] = *(const uint4*)(rowbase + kb + 16);
    return r;
}

#define WMMA_BF16(a, b, c) \
    __builtin_amdgcn_wmma_f32_16x16x32_bf16(false, (a), false, (b), (short)0, (c), false, false)

// ---------------------------------------------------------------------------
// Kernel 1: one-shot f32 -> bf16 of X (16.8M), W_ih (2.1M), fc_w (0.5M)
// ---------------------------------------------------------------------------
__global__ __launch_bounds__(256) void cvt_kernel(
        const float4* __restrict__ x, unsigned short* __restrict__ ox, int nx4,
        const float4* __restrict__ w, unsigned short* __restrict__ ow, int nw4,
        const float4* __restrict__ f, unsigned short* __restrict__ of, int nf4) {
    int i = blockIdx.x * blockDim.x + threadIdx.x;
    if (i < nx4) {
        float4 v = x[i];
        unsigned short t[4] = {f2bf(v.x), f2bf(v.y), f2bf(v.z), f2bf(v.w)};
        *(uint2*)(ox + 4 * (size_t)i) = *(const uint2*)t;
    }
    if (i < nw4) {
        float4 v = w[i];
        unsigned short t[4] = {f2bf(v.x), f2bf(v.y), f2bf(v.z), f2bf(v.w)};
        *(uint2*)(ow + 4 * (size_t)i) = *(const uint2*)t;
    }
    if (i < nf4) {
        float4 v = f[i];
        unsigned short t[4] = {f2bf(v.x), f2bf(v.y), f2bf(v.z), f2bf(v.w)};
        *(uint2*)(of + 4 * (size_t)i) = *(const uint2*)t;
    }
}

// ---------------------------------------------------------------------------
// Kernel 2: state_part[b,n] = h0[b,:] . W_hh[n,:] + b_ih[n] + b_hh[n]
//  [128,4096], K=1024 -> ~1 GFLOP, f32 VALU (too small for WMMA to matter)
// ---------------------------------------------------------------------------
__global__ __launch_bounds__(256) void state_kernel(const float* __restrict__ h0,
                                                    const float* __restrict__ W_hh,
                                                    const float* __restrict__ b_ih,
                                                    const float* __restrict__ b_hh,
                                                    float* __restrict__ state) {
    int idx = blockIdx.x * blockDim.x + threadIdx.x;
    int b = idx >> 12;
    int n = idx & (DIM_G - 1);
    const float4* hp = (const float4*)(h0 + (size_t)b * DIM_H);
    const float4* wp = (const float4*)(W_hh + (size_t)n * DIM_H);
    float acc = 0.0f;
    #pragma unroll 8
    for (int k = 0; k < DIM_H / 4; ++k) {
        float4 a = hp[k], w = wp[k];
        acc += a.x * w.x + a.y * w.y + a.z * w.z + a.w * w.w;
    }
    state[idx] = acc + b_ih[n] + b_hh[n];
}

// ---------------------------------------------------------------------------
// Kernel 3: gates GEMM (32768 x 4096, K=512) + fused LSTM epilogue -> h (bf16)
//  Block = 128 rows x 64 h-cols (=256 gate cols: 4 gates x 64).
//  8 waves as 4(M) x 2(N); each wave: 32 rows x 32 h-cols -> 16 acc tiles
//  covering all 4 gates, so i/f/g/o combine entirely in registers.
//  Double-buffered LDS fed by async global->LDS b128 copies (ASYNCcnt).
// ---------------------------------------------------------------------------
__global__ __launch_bounds__(256) void gates_kernel(
        const unsigned short* __restrict__ Xbf,   // [32768,512] bf16
        const float* __restrict__ c0,             // [128,1024]
        const float* __restrict__ state,          // [128,4096]
        const unsigned short* __restrict__ Wih,   // [4096,512] bf16
        unsigned short* __restrict__ hbuf) {      // [32768,1024] bf16
    __shared__ unsigned short ldsA[2][128 * LDA];  // X rows
    __shared__ unsigned short ldsB[2][256 * LDA];  // 4 gates x 64 cols of W_ih

    const int j0    = blockIdx.x * 64;    // h-column block
    const int r0    = blockIdx.y * 128;   // row block
    const int tid   = threadIdx.x;
    const int lane  = tid & 31;
    const int wave  = tid >> 5;
    const int waveM = wave & 3;           // 4 row groups of 32
    const int waveN = wave >> 2;          // 2 col groups of 32
    const int bIdx  = r0 >> 8;            // one batch index per block (128 | 256)

    v8f acc[2][4][2];
    #pragma unroll
    for (int m = 0; m < 2; ++m)
        #pragma unroll
        for (int g = 0; g < 4; ++g)
            #pragma unroll
            for (int c = 0; c < 2; ++c)
                #pragma unroll
                for (int e = 0; e < 8; ++e) acc[m][g][c][e] = 0.0f;

    // tile staging: 6 b128 copies per thread per k-step
    const int ldRow = tid >> 1;           // 0..127
    const int ldOff = (tid & 1) * 16;     // 0 or 16
    const int b0row = ldRow;              // ldsB rows 0..127   (gates 0,1)
    const int b1row = ldRow + 128;        // ldsB rows 128..255 (gates 2,3)
    const unsigned short* gA  = Xbf + (size_t)(r0 + ldRow) * DIM_I + ldOff;
    const unsigned short* gB0 = Wih + (size_t)((b0row >> 6) * DIM_H + j0 + (b0row & 63)) * DIM_I + ldOff;
    const unsigned short* gB1 = Wih + (size_t)((b1row >> 6) * DIM_H + j0 + (b1row & 63)) * DIM_I + ldOff;

    auto issue = [&](int p, int k0) {
        unsigned short* la = &ldsA[p][ldRow * LDA + ldOff];
        cp_b128(gA + k0,     la);
        cp_b128(gA + k0 + 8, la + 8);
        unsigned short* lb0 = &ldsB[p][b0row * LDA + ldOff];
        cp_b128(gB0 + k0,     lb0);
        cp_b128(gB0 + k0 + 8, lb0 + 8);
        unsigned short* lb1 = &ldsB[p][b1row * LDA + ldOff];
        cp_b128(gB1 + k0,     lb1);
        cp_b128(gB1 + k0 + 8, lb1 + 8);
    };

    const int laneN = lane & 15;
    const int kb    = (lane < 16) ? 0 : 8;
    const int NK    = DIM_I / 32;         // 16

    issue(0, 0);
    for (int ks = 0; ks < NK; ++ks) {
        const int p = ks & 1;
        if (ks + 1 < NK) {
            __syncthreads();              // everyone done reading buf p^1
            issue(p ^ 1, (ks + 1) * 32);
            TILE_WAIT(6);                 // wait for stage ks (last 6 are ks+1)
        } else {
            TILE_WAIT(0);
        }
        __syncthreads();                  // all waves' stage-ks tiles visible

        bf16x16 af0 = load_frag(&ldsA[p][(waveM * 32 +      laneN) * LDA], kb);
        bf16x16 af1 = load_frag(&ldsA[p][(waveM * 32 + 16 + laneN) * LDA], kb);
        #pragma unroll
        for (int g = 0; g < 4; ++g) {
            #pragma unroll
            for (int c = 0; c < 2; ++c) {
                bf16x16 bf = load_frag(
                    &ldsB[p][(g * 64 + waveN * 32 + c * 16 + laneN) * LDA], kb);
                acc[0][g][c] = WMMA_BF16(af0, bf, acc[0][g][c]);
                acc[1][g][c] = WMMA_BF16(af1, bf, acc[1][g][c]);
            }
        }
    }

    // Fused LSTM epilogue: h = sigmoid(o) * tanh(sigmoid(f)*c0 + sigmoid(i)*tanh(g))
    const int hiOff = (lane >> 4) * 8;    // C/D layout: lanes>=16 hold M+8
    int   col[2];
    float sI[2], sF[2], sG[2], sO[2], c0v[2];
    #pragma unroll
    for (int c = 0; c < 2; ++c) {
        col[c] = j0 + waveN * 32 + c * 16 + laneN;
        const float* sp = state + (size_t)bIdx * DIM_G;
        sI[c]  = sp[col[c]];
        sF[c]  = sp[DIM_H + col[c]];
        sG[c]  = sp[2 * DIM_H + col[c]];
        sO[c]  = sp[3 * DIM_H + col[c]];
        c0v[c] = c0[(size_t)bIdx * DIM_H + col[c]];
    }
    #pragma unroll
    for (int m = 0; m < 2; ++m) {
        #pragma unroll
        for (int r = 0; r < 8; ++r) {
            int row = r0 + waveM * 32 + m * 16 + r + hiOff;
            #pragma unroll
            for (int c = 0; c < 2; ++c) {
                float gi = sigmoidf_(acc[m][0][c][r] + sI[c]);
                float gf = sigmoidf_(acc[m][1][c][r] + sF[c]);
                float gg = TANHF    (acc[m][2][c][r] + sG[c]);
                float go = sigmoidf_(acc[m][3][c][r] + sO[c]);
                float cc = gf * c0v[c] + gi * gg;
                float hv = go * TANHF(cc);
                hbuf[(size_t)row * DIM_H + col[c]] = f2bf(hv);
            }
        }
    }
}

// ---------------------------------------------------------------------------
// Kernel 4: out = h @ fc_w^T + fc_b  (32768 x 512, K=1024), written [T,B,I]
//  Block = 128 rows x 128 cols; 8 waves as 4(M) x 2(N): 32 rows x 64 cols each.
// ---------------------------------------------------------------------------
__global__ __launch_bounds__(256) void out_kernel(
        const unsigned short* __restrict__ hbuf,  // [32768,1024] bf16
        const unsigned short* __restrict__ fcw,   // [512,1024] bf16
        const float* __restrict__ fcb,            // [512]
        float* __restrict__ out) {                // [256,128,512]
    __shared__ unsigned short ldsA[2][128 * LDA];
    __shared__ unsigned short ldsB[2][128 * LDA];

    const int n0    = blockIdx.x * 128;
    const int r0    = blockIdx.y * 128;
    const int tid   = threadIdx.x;
    const int lane  = tid & 31;
    const int wave  = tid >> 5;
    const int waveM = wave & 3;
    const int waveN = wave >> 2;
    const int bIdx  = r0 >> 8;

    v8f acc[2][4];
    #pragma unroll
    for (int m = 0; m < 2; ++m)
        #pragma unroll
        for (int t = 0; t < 4; ++t)
            #pragma unroll
            for (int e = 0; e < 8; ++e) acc[m][t][e] = 0.0f;

    const int ldRow = tid >> 1;
    const int ldOff = (tid & 1) * 16;
    const unsigned short* gA = hbuf + (size_t)(r0 + ldRow) * DIM_H + ldOff;
    const unsigned short* gB = fcw  + (size_t)(n0 + ldRow) * DIM_H + ldOff;

    auto issue = [&](int p, int k0) {
        unsigned short* la = &ldsA[p][ldRow * LDA + ldOff];
        cp_b128(gA + k0,     la);
        cp_b128(gA + k0 + 8, la + 8);
        unsigned short* lb = &ldsB[p][ldRow * LDA + ldOff];
        cp_b128(gB + k0,     lb);
        cp_b128(gB + k0 + 8, lb + 8);
    };

    const int laneN = lane & 15;
    const int kb    = (lane < 16) ? 0 : 8;
    const int NK    = DIM_H / 32;         // 32

    issue(0, 0);
    for (int ks = 0; ks < NK; ++ks) {
        const int p = ks & 1;
        if (ks + 1 < NK) {
            __syncthreads();
            issue(p ^ 1, (ks + 1) * 32);
            TILE_WAIT(4);
        } else {
            TILE_WAIT(0);
        }
        __syncthreads();

        bf16x16 af0 = load_frag(&ldsA[p][(waveM * 32 +      laneN) * LDA], kb);
        bf16x16 af1 = load_frag(&ldsA[p][(waveM * 32 + 16 + laneN) * LDA], kb);
        #pragma unroll
        for (int t = 0; t < 4; ++t) {
            bf16x16 bf = load_frag(&ldsB[p][(waveN * 64 + t * 16 + laneN) * LDA], kb);
            acc[0][t] = WMMA_BF16(af0, bf, acc[0][t]);
            acc[1][t] = WMMA_BF16(af1, bf, acc[1][t]);
        }
    }

    // Epilogue: bias add + scatter f32 into [T,B,I] = t*65536 + b*512 + n
    const int hiOff = (lane >> 4) * 8;
    float bias[4];
    #pragma unroll
    for (int t = 0; t < 4; ++t) bias[t] = fcb[n0 + waveN * 64 + t * 16 + laneN];

    #pragma unroll
    for (int m = 0; m < 2; ++m) {
        #pragma unroll
        for (int r = 0; r < 8; ++r) {
            int row  = r0 + waveM * 32 + m * 16 + r + hiOff;
            int tIdx = row & (DIM_T - 1);
            float* orow = out + (size_t)tIdx * DIM_B * DIM_I + (size_t)bIdx * DIM_I;
            #pragma unroll
            for (int t = 0; t < 4; ++t) {
                orow[n0 + waveN * 64 + t * 16 + laneN] = acc[m][t][r] + bias[t];
            }
        }
    }
}

// ---------------------------------------------------------------------------
// Launch
// ---------------------------------------------------------------------------
extern "C" void kernel_launch(void* const* d_in, const int* in_sizes, int n_in,
                              void* d_out, int out_size, void* d_ws, size_t ws_size,
                              hipStream_t stream) {
    const float* X    = (const float*)d_in[0];   // [128,256,512]
    const float* h0   = (const float*)d_in[1];   // [128,1024]
    const float* c0   = (const float*)d_in[2];   // [128,1024]
    const float* W_ih = (const float*)d_in[3];   // [4096,512]
    const float* W_hh = (const float*)d_in[4];   // [4096,1024]
    const float* b_ih = (const float*)d_in[5];   // [4096]
    const float* b_hh = (const float*)d_in[6];   // [4096]
    const float* fc_w = (const float*)d_in[7];   // [512,1024]
    const float* fc_b = (const float*)d_in[8];   // [512]
    float* out = (float*)d_out;                  // [256,128,512]

    char* ws = (char*)d_ws;
    unsigned short* Xbf   = (unsigned short*)(ws);                  // 32 MB
    unsigned short* Wihbf = (unsigned short*)(ws + (32ull << 20));  //  4 MB
    unsigned short* fcwbf = (unsigned short*)(ws + (36ull << 20));  //  1 MB
    float*          state = (float*)         (ws + (40ull << 20));  //  2 MB
    unsigned short* hbuf  = (unsigned short*)(ws + (48ull << 20));  // 64 MB

    const int nX4 = ROWS * DIM_I / 4;    // 4194304
    const int nW4 = DIM_G * DIM_I / 4;   //  524288
    const int nF4 = DIM_I * DIM_H / 4;   //  131072
    cvt_kernel<<<(nX4 + 255) / 256, 256, 0, stream>>>(
        (const float4*)X, Xbf, nX4,
        (const float4*)W_ih, Wihbf, nW4,
        (const float4*)fc_w, fcwbf, nF4);

    state_kernel<<<(DIM_B * DIM_G) / 256, 256, 0, stream>>>(h0, W_hh, b_ih, b_hh, state);

    gates_kernel<<<dim3(DIM_H / 64, ROWS / 128), 256, 0, stream>>>(Xbf, c0, state, Wihbf, hbuf);

    out_kernel<<<dim3(DIM_I / 128, ROWS / 128), 256, 0, stream>>>(hbuf, fcwbf, fc_b, out);
}
